// ImprovedGATLayer_66907000537680
// MI455X (gfx1250) — compile-verified, matrix operand order checked by reading, not run on hardware
//
#include <hip/hip_runtime.h>
#include <math.h>

typedef __attribute__((ext_vector_type(16))) _Float16 v16h;
typedef __attribute__((ext_vector_type(8)))  _Float16 v8h;
typedef __attribute__((ext_vector_type(8)))  float    v8f;
typedef __attribute__((ext_vector_type(4)))  float    v4f;

#define B_   8
#define N_   2048
#define FIN_ 64
#define FOUT_ 64
#define H_   4
#define BH_  (B_*H_)

static __device__ __forceinline__ float warp_red_max(float v) {
#pragma unroll
  for (int m = 16; m >= 1; m >>= 1) v = fmaxf(v, __shfl_xor(v, m, 32));
  return v;
}
static __device__ __forceinline__ float warp_red_sum(float v) {
#pragma unroll
  for (int m = 16; m >= 1; m >>= 1) v += __shfl_xor(v, m, 32);
  return v;
}

// ---------------------------------------------------------------------------
// k0: convert W (64x256 f32, row-major) -> Wt (f16, [col][k], 256x64)
// ---------------------------------------------------------------------------
__global__ __launch_bounds__(256, 4)
void k0_wcvt(const float* __restrict__ W, _Float16* __restrict__ Wt) {
  const int c = blockIdx.x * blockDim.x + threadIdx.x;  // 0..255
#pragma unroll
  for (int k = 0; k < FIN_; ++k)
    Wt[(size_t)c * FIN_ + k] = (_Float16)W[(size_t)k * (H_ * FOUT_) + c];
}

// ---------------------------------------------------------------------------
// k1: h = x @ W  (16384x64 * 64x256), WMMA f16 -> f32 accum.
// One wave = 16 rows x 64 cols (one head), K=64 (2 steps of 32).
// All fragment loads are aligned b128s. Epilogue stores h f16 TRANSPOSED:
//   h_t[((b*H+h)*FOUT + f)*N + n]
// ---------------------------------------------------------------------------
__global__ __launch_bounds__(256, 2)
void k1_xw_gemm(const float* __restrict__ x, const _Float16* __restrict__ Wt,
                _Float16* __restrict__ h_t) {
  const int lane = threadIdx.x & 31;
  const int wave = threadIdx.x >> 5;
  const int wg   = blockIdx.x * 8 + wave;     // 0..4095
  const int rowTile = wg >> 2;                // 0..1023
  const int g = wg & 3;                       // head / 64-col group
  const int half = lane >> 4;
  const int mrow = lane & 15;
  const int kbase = half * 8;
  const int row0 = rowTile * 16;

  v8f acc[4] = {};

#pragma unroll
  for (int ks = 0; ks < 2; ++ks) {
    const int k0 = ks * 32;
    // A fragment: 16x32 f16 from x rows; two 16B-aligned float4 pairs
    v16h afrag;
    const float* xp = x + (size_t)(row0 + mrow) * FIN_ + k0 + kbase;
    const v4f xa = ((const v4f*)xp)[0];
    const v4f xb = ((const v4f*)xp)[1];
    const v4f xc = ((const v4f*)(xp + 16))[0];
    const v4f xd = ((const v4f*)(xp + 16))[1];
#pragma unroll
    for (int i = 0; i < 4; ++i) {
      afrag[i]      = (_Float16)xa[i];
      afrag[4 + i]  = (_Float16)xb[i];
      afrag[8 + i]  = (_Float16)xc[i];
      afrag[12 + i] = (_Float16)xd[i];
    }
#pragma unroll
    for (int t = 0; t < 4; ++t) {
      const int col = g * 64 + t * 16 + mrow;
      const _Float16* wp = Wt + (size_t)col * FIN_ + k0 + kbase;
      const v8h lo = *(const v8h*)(wp);
      const v8h hi = *(const v8h*)(wp + 16);
      v16h bfrag;
#pragma unroll
      for (int q = 0; q < 8; ++q) { bfrag[q] = lo[q]; bfrag[q + 8] = hi[q]; }
      acc[t] = __builtin_amdgcn_wmma_f32_16x16x32_f16(
          false, afrag, false, bfrag, (short)0, acc[t], false, false);
    }
  }

  // store transposed f16
  const int rowg = row0 + (half ? 8 : 0);
#pragma unroll
  for (int t = 0; t < 4; ++t) {
#pragma unroll
    for (int r = 0; r < 8; ++r) {
      const int row = rowg + r;           // global row in (B*N)
      const int b = row >> 11;
      const int n = row & (N_ - 1);
      const int f = t * 16 + mrow;
      h_t[(((size_t)(b * H_ + g) * FOUT_) + f) * N_ + n] = (_Float16)acc[t][r];
    }
  }
}

// ---------------------------------------------------------------------------
// k1b: s_src[b,h,n] = <h[b,h,n,:], a_src[h]>, s_dst likewise
// ---------------------------------------------------------------------------
__global__ __launch_bounds__(256, 4)
void k1b_scores(const _Float16* __restrict__ h_t, const float* __restrict__ a,
                float* __restrict__ s_src, float* __restrict__ s_dst) {
  const int idx = blockIdx.x * blockDim.x + threadIdx.x;  // 0..BH*N-1
  const int n = idx & (N_ - 1);
  const int bh = idx >> 11;
  const int h = bh & (H_ - 1);
  float ss = 0.f, sd = 0.f;
  const _Float16* hp = h_t + (size_t)bh * FOUT_ * N_ + n;
#pragma unroll
  for (int f = 0; f < FOUT_; ++f) {
    float v = (float)hp[(size_t)f * N_];
    ss += v * a[h * 2 * FOUT_ + f];
    sd += v * a[h * 2 * FOUT_ + FOUT_ + f];
  }
  s_src[idx] = ss;
  s_dst[idx] = sd;
}

// ---------------------------------------------------------------------------
// k2a: softmax stats per row (max, sum-of-exp), one wave per (b,h,i)
// ---------------------------------------------------------------------------
__global__ __launch_bounds__(256, 2)
void k2a_stats(const float* __restrict__ s_src, const float* __restrict__ s_dst,
               const int* __restrict__ adj, float* __restrict__ m_stat,
               float* __restrict__ l_stat) {
  const int lane = threadIdx.x & 31;
  const int wave = threadIdx.x >> 5;
  const int row = blockIdx.x * 8 + wave;  // 0..BH*N-1
  const int i = row & (N_ - 1);
  const int bh = row >> 11;
  const float si = s_src[row];
  const float* sdp = s_dst + (size_t)bh * N_;
  const int* ap = adj + (size_t)i * N_;

  float mx = -3.0e38f;
  for (int j = lane; j < N_; j += 32) {
    float e = si + sdp[j];
    e = e > 0.f ? e : 0.2f * e;
    if (ap[j] != 0) mx = fmaxf(mx, e);
  }
  mx = warp_red_max(mx);

  float sum = 0.f;
  for (int j = lane; j < N_; j += 32) {
    float e = si + sdp[j];
    e = e > 0.f ? e : 0.2f * e;
    if (ap[j] != 0) sum += __expf(e - mx);
  }
  sum = warp_red_sum(sum);

  if (lane == 0) { m_stat[row] = mx; l_stat[row] = sum; }
}

// ---------------------------------------------------------------------------
// k2b: h_out[b,h,i,:] = alpha @ h   (fused softmax-normalize + WMMA)
// Block = 4 waves; all waves share one 16-row query tile. The 16x32 int
// adjacency tile (2 KiB) is staged via CDNA5 async DMA: each of the 128
// threads issues one global_load_async_to_lds_b128 (ASYNCcnt-tracked),
// then s_wait_asynccnt 0 + barrier. Each wave owns one (b,h): builds the
// f16 alpha A-fragment on the fly and runs 4 WMMAs per 32-column step.
// ---------------------------------------------------------------------------
__global__ __launch_bounds__(128, 2)
void k2b_attn(const float* __restrict__ s_src, const float* __restrict__ s_dst,
              const float* __restrict__ m_stat, const float* __restrict__ l_stat,
              const int* __restrict__ adj, const _Float16* __restrict__ h_t,
              float* __restrict__ h_out) {
  __shared__ int s_adj[16][32];
  const int lane = threadIdx.x & 31;
  const int wave = threadIdx.x >> 5;
  const int iTile = blockIdx.x;             // 0..127
  const int bh = blockIdx.y * 4 + wave;     // 0..31
  const int i0 = iTile * 16;
  const int half = lane >> 4;
  const int mrow = lane & 15;
  const int kbase = half * 8;

  const int rowIdx = bh * N_ + i0 + mrow;
  const float si   = s_src[rowIdx];
  const float mval = m_stat[rowIdx];
  const float invl = 1.0f / l_stat[rowIdx];

  v8f acc[4] = {};
  const _Float16* hbase = h_t + (size_t)bh * FOUT_ * N_;

  // per-thread async staging coordinates: 128 threads x 16B = 16x32 ints
  const int rr = threadIdx.x >> 3;          // 0..15
  const int cc = (threadIdx.x & 7) * 4;     // 0,4,...,28
  // LDS byte address of this thread's 16B slot (flat addr low 32 bits == LDS addr)
  const unsigned ldsa =
      (unsigned)(size_t)(&s_adj[0][0]) + (unsigned)((rr * 32 + cc) << 2);

  for (int jb = 0; jb < N_; jb += 32) {
    __syncthreads();  // previous tile fully consumed
    {
      const unsigned long long ga =
          (unsigned long long)(size_t)(adj + (size_t)(i0 + rr) * N_ + jb + cc);
      asm volatile("global_load_async_to_lds_b128 %0, %1, off"
                   :: "v"(ldsa), "v"(ga) : "memory");
      asm volatile("s_wait_asynccnt 0x0" ::: "memory");
    }
    __syncthreads();  // tile visible to all 4 waves

    // per-lane column score for this block, distributed via bpermute
    const float sdj = s_dst[(size_t)bh * N_ + jb + lane];

    // alpha A-fragment (16x32 f16), already softmax-normalized
    v16h afrag;
#pragma unroll
    for (int q = 0; q < 16; ++q) {
      const int kl = ((q < 8) ? 0 : 16) + kbase + (q & 7);
      const float sj = __shfl(sdj, kl, 32);
      float e = si + sj;
      e = e > 0.f ? e : 0.2f * e;
      const float alpha =
          (s_adj[mrow][kl] != 0) ? __expf(e - mval) * invl : 0.f;
      afrag[q] = (_Float16)alpha;
    }

#pragma unroll
    for (int t = 0; t < 4; ++t) {
      const _Float16* hp =
          hbase + (size_t)(t * 16 + mrow) * N_ + jb + kbase;
      const v8h lo = *(const v8h*)(hp);
      const v8h hi = *(const v8h*)(hp + 16);
      v16h bfrag;
#pragma unroll
      for (int q = 0; q < 8; ++q) { bfrag[q] = lo[q]; bfrag[q + 8] = hi[q]; }
      acc[t] = __builtin_amdgcn_wmma_f32_16x16x32_f16(
          false, afrag, false, bfrag, (short)0, acc[t], false, false);
    }
    if (lane == 0 && jb + 32 < N_)
      __builtin_prefetch(adj + (size_t)i0 * N_ + jb + 32, 0, 1);
  }

  // store D tiles: row = i0 + r + half*8, f = 16t + mrow (coalesced in f)
  const int rg = i0 + (half ? 8 : 0);
#pragma unroll
  for (int t = 0; t < 4; ++t)
#pragma unroll
    for (int r = 0; r < 8; ++r)
      h_out[((size_t)bh * N_ + rg + r) * FOUT_ + t * 16 + mrow] = acc[t][r];
}

// ---------------------------------------------------------------------------
// k3: head mean + residual + LayerNorm; one wave per (b,n) row
// ---------------------------------------------------------------------------
__global__ __launch_bounds__(256, 4)
void k3_ln(const float* __restrict__ x, const float* __restrict__ h_out,
           const float* __restrict__ gamma, const float* __restrict__ beta,
           float* __restrict__ out) {
  const int lane = threadIdx.x & 31;
  const int wave = threadIdx.x >> 5;
  const int row = blockIdx.x * 8 + wave;  // b*N + n
  const int b = row >> 11;
  const int n = row & (N_ - 1);
  float y[2];
#pragma unroll
  for (int q = 0; q < 2; ++q) {
    const int f = lane + q * 32;
    float s = 0.f;
#pragma unroll
    for (int h = 0; h < H_; ++h)
      s += h_out[(((size_t)(b * H_ + h) * N_) + n) * FOUT_ + f];
    y[q] = s * (1.0f / H_) + x[(size_t)row * FOUT_ + f];
  }
  float s1 = y[0] + y[1];
  float s2 = y[0] * y[0] + y[1] * y[1];
  s1 = warp_red_sum(s1);
  s2 = warp_red_sum(s2);
  const float mu = s1 * (1.0f / FOUT_);
  const float var = s2 * (1.0f / FOUT_) - mu * mu;
  const float rstd = rsqrtf(var + 1e-5f);
#pragma unroll
  for (int q = 0; q < 2; ++q) {
    const int f = lane + q * 32;
    out[(size_t)row * FOUT_ + f] = (y[q] - mu) * rstd * gamma[f] + beta[f];
  }
}

// ---------------------------------------------------------------------------
extern "C" void kernel_launch(void* const* d_in, const int* in_sizes, int n_in,
                              void* d_out, int out_size, void* d_ws,
                              size_t ws_size, hipStream_t stream) {
  const float* x     = (const float*)d_in[0];
  const int*   adj   = (const int*)d_in[1];
  const float* W     = (const float*)d_in[2];
  const float* a     = (const float*)d_in[3];
  const float* gamma = (const float*)d_in[4];
  const float* beta  = (const float*)d_in[5];
  float* out = (float*)d_out;

  char* ws = (char*)d_ws;
  _Float16* h_t = (_Float16*)(ws);                              // 8 MiB
  float* s_src  = (float*)(ws + (8u << 20));                    // 256 KiB
  float* s_dst  = (float*)(ws + (8u << 20) + (256u << 10));     // 256 KiB
  float* m_stat = (float*)(ws + (8u << 20) + (512u << 10));     // 256 KiB
  float* l_stat = (float*)(ws + (8u << 20) + (768u << 10));     // 256 KiB
  float* h_out  = (float*)(ws + (9u << 20));                    // 16 MiB
  _Float16* Wt  = (_Float16*)(ws + (25u << 20));                // 32 KiB

  k0_wcvt<<<1, 256, 0, stream>>>(W, Wt);
  k1_xw_gemm<<<512, 256, 0, stream>>>(x, Wt, h_t);
  k1b_scores<<<(BH_ * N_) / 256, 256, 0, stream>>>(h_t, a, s_src, s_dst);
  k2a_stats<<<(BH_ * N_) / 8, 256, 0, stream>>>(s_src, s_dst, adj, m_stat,
                                                l_stat);
  dim3 g2b(128, 8);
  k2b_attn<<<g2b, 128, 0, stream>>>(s_src, s_dst, m_stat, l_stat, adj, h_t,
                                    h_out);
  k3_ln<<<(B_ * N_) / 8, 256, 0, stream>>>(x, h_out, gamma, beta, out);
}
